// MODEL_68547678044706
// MI455X (gfx1250) — matrix operand order repaired and checked by reading, not compiled, not run on hardware
//
#include <hip/hip_runtime.h>
#include <hip/hip_bf16.h>

// B=256, T=250, D=700, H=256, O=20, SKIP=21, VTH=1, DECAY=0.5
// prep:  W1 -> bf16 [256,704] zero-padded, W2 -> bf16 [256,256]   (workspace)
// ff:    H1 = LN(x @ W1^T) big parallel GEMM, v_wmma_f32_16x16x32_bf16
// scan:  sequential LIF scan; M1/M2 resident in 256 KB dynamic LDS as bf16;
//        16 blocks (one per WGP) x 16-row B-slices.

typedef __attribute__((ext_vector_type(16))) __bf16 v16bf;
typedef __attribute__((ext_vector_type(8)))  float  v8f;

__device__ __forceinline__ v8f wmma_bf16(v16bf a, v16bf b, v8f c) {
    return __builtin_amdgcn_wmma_f32_16x16x32_bf16(
        false, a, false, b, (short)0, c, false, false);
}

// A fragment (16x32 bf16, MxK) from LDS row-major bf16 tile.
// lanes 0-15: M=lane, K=0..15; lanes 16-31: M=lane-16, K=16..31.
__device__ __forceinline__ v16bf load_a_lds(const __bf16* tile, int ld, int k0, int lane) {
    int m  = lane & 15;
    int kh = (lane >> 4) * 16;
    const __bf16* rowp = tile + m * ld + k0;
    v16bf a;
#pragma unroll
    for (int v = 0; v < 8; ++v) {
        int kb = kh + 2 * v;
        a[2 * v]     = rowp[kb];
        a[2 * v + 1] = rowp[kb + 1];
    }
    return a;
}

// B fragment (32x16, KxN) with B[k][n] = W[n][k]; W row-major bf16 [N][ld].
// lanes 0-15: N=lane, K=0..15; lanes 16-31: N=lane-16, K=16..31 -> 16 contiguous bf16 per lane.
__device__ __forceinline__ v16bf load_b_bf16(const __bf16* W, int ld, int n0, int k0, int lane) {
    int n  = lane & 15;
    int kh = (lane >> 4) * 16;
    const __bf16* row = W + (size_t)(n0 + n) * ld + k0 + kh;
    v16bf b;
#pragma unroll
    for (int v = 0; v < 8; ++v) {
        b[2 * v]     = row[2 * v];
        b[2 * v + 1] = row[2 * v + 1];
    }
    return b;
}

// ---------------- prep: convert weights to bf16 (W1 zero-padded to K=704) ----------------
__global__ __launch_bounds__(512) void prep_weights_kernel(
    const float* __restrict__ W1, const float* __restrict__ W2,
    __bf16* __restrict__ W1bf, __bf16* __restrict__ W2bf)
{
    int i = blockIdx.x * 512 + threadIdx.x;
    if (i < 256 * 704) {
        int n = i / 704, k = i % 704;
        W1bf[i] = (k < 700) ? (__bf16)W1[n * 700 + k] : (__bf16)0.0f;
    }
    if (i < 256 * 256) W2bf[i] = (__bf16)W2[i];
}

// ---------------- Phase 1: H1[b*T+t][h] = LN(x @ W1^T) ----------------
// grid = 4000 row-tiles of 16, block = 512 (16 waves; wave w owns column tile w).
__global__ __launch_bounds__(512) void ff_gemm_ln_kernel(
    const float* __restrict__ X, const __bf16* __restrict__ W1bf,
    const float* __restrict__ g1, const float* __restrict__ b1,
    float* __restrict__ H1)
{
    const int tid  = threadIdx.x;
    const int lane = tid & 31;
    const int wave = tid >> 5;        // column tile
    const int r0   = blockIdx.x * 16; // row tile over B*T = 64000

    __shared__ __bf16 aTile[2][16 * 32];
    __shared__ float  rowsum[16], rowsumsq[16];

    const int sm = tid >> 5, sk = tid & 31; // staging coords (one elem/thread)
    // preload K-tile 0
    {
        float xv = (sk < 700) ? X[(size_t)(r0 + sm) * 700 + sk] : 0.0f;
        aTile[0][sm * 32 + sk] = (__bf16)xv;
    }
    __syncthreads();

    v8f acc = {};
    for (int it = 0; it < 22; ++it) {
        const int k0  = it * 32;
        const int cur = it & 1;
        if (it + 1 < 22) {
            int k = k0 + 32 + sk;
            float xv = (k < 700) ? X[(size_t)(r0 + sm) * 700 + k] : 0.0f;
            aTile[cur ^ 1][sm * 32 + sk] = (__bf16)xv;
        }
        v16bf a = load_a_lds(aTile[cur], 32, 0, lane);
        v16bf b = load_b_bf16(W1bf, 704, wave * 16, k0, lane);
        acc = wmma_bf16(a, b, acc);
        __syncthreads();
    }

    if (tid < 16) { rowsum[tid] = 0.0f; rowsumsq[tid] = 0.0f; }
    __syncthreads();
#pragma unroll
    for (int v = 0; v < 8; ++v) {
        float p = acc[v], q = acc[v] * acc[v];
#pragma unroll
        for (int m = 1; m < 16; m <<= 1) {
            p += __shfl_xor(p, m, 32);
            q += __shfl_xor(q, m, 32);
        }
        int row = v + ((lane >> 4) << 3);
        if ((lane & 15) == 0) {
            atomicAdd(&rowsum[row], p);
            atomicAdd(&rowsumsq[row], q);
        }
    }
    __syncthreads();

    const int col = wave * 16 + (lane & 15);
    const float gg = g1[col], bb = b1[col];
#pragma unroll
    for (int v = 0; v < 8; ++v) {
        int row = v + ((lane >> 4) << 3);
        float mean = rowsum[row] * (1.0f / 256.0f);
        float var  = rowsumsq[row] * (1.0f / 256.0f) - mean * mean;
        float y = (acc[v] - mean) * rsqrtf(var + 1e-5f) * gg + bb;
        H1[(size_t)(r0 + row) * 256 + col] = y;
    }
}

// ---------------- Phase 2: sequential LIF scan ----------------
// grid = 16, block = 512 (16 waves). 256 KB dynamic LDS holds M1,M2 as bf16.
__global__ __launch_bounds__(512) void scan_kernel(
    const float* __restrict__ H1,
    const float* __restrict__ M1, const float* __restrict__ sp1,
    const __bf16* __restrict__ W2bf, const float* __restrict__ g2,
    const float* __restrict__ b2, const float* __restrict__ M2,
    const float* __restrict__ sp2, const float* __restrict__ Wout,
    float* __restrict__ out)
{
    const int tid  = threadIdx.x;
    const int lane = tid & 31;
    const int wave = tid >> 5;
    const int b0   = blockIdx.x * 16;

    extern __shared__ __align__(16) char dynsm[];
    __bf16* m1l = (__bf16*)dynsm;          // 256x256 bf16 = 128 KB
    __bf16* m2l = m1l + 256 * 256;         // 128 KB

    __shared__ __bf16 astage[16 * 256];            // A operand staging
    __shared__ unsigned short hist1[21 * 256];     // spike history bitmasks
    __shared__ unsigned short hist2[21 * 256];
    __shared__ float w1s[21], w2s[21];
    __shared__ float rowsum[16], rowsumsq[16];
    __shared__ float s2dump[16 * 256];

    // stage recurrent weights into LDS (one-time, off the sequential path)
    for (int i = tid; i < 256 * 256; i += 512) {
        m1l[i] = (__bf16)M1[i];
        m2l[i] = (__bf16)M2[i];
    }

    float v1r[8], v2r[8], s2r[8];
#pragma unroll
    for (int v = 0; v < 8; ++v) { v1r[v] = 0.0f; v2r[v] = 0.0f; s2r[v] = 0.0f; }

    for (int i = tid; i < 21 * 256; i += 512) { hist1[i] = 0; hist2[i] = 0; }
    if (tid == 0) {
        float mx = -1e30f, sum = 0.0f;
        for (int k = 0; k < 21; ++k) mx = fmaxf(mx, sp1[k]);
        for (int k = 0; k < 21; ++k) { float e = __expf(sp1[k] - mx); w1s[k] = e; sum += e; }
        for (int k = 0; k < 21; ++k) w1s[k] /= sum;
        mx = -1e30f; sum = 0.0f;
        for (int k = 0; k < 21; ++k) mx = fmaxf(mx, sp2[k]);
        for (int k = 0; k < 21; ++k) { float e = __expf(sp2[k] - mx); w2s[k] = e; sum += e; }
        for (int k = 0; k < 21; ++k) w2s[k] /= sum;
    }
    __syncthreads();

    const int col = wave * 16 + (lane & 15);
    const float gg2 = g2[col], bb2 = b2[col];

    for (int t = 0; t < 250; ++t) {
        const int rhead = (21 - (t % 21)) % 21; // slot holding spike_{t-1}
        const int wslot = 20 - (t % 21);        // slot receiving spike_t

        // prefetch next timestep's H1 rows (global_prefetch_b8)
        if (t + 1 < 250) {
#pragma unroll
            for (int v = 0; v < 8; ++v) {
                int row = v + ((lane >> 4) << 3);
                __builtin_prefetch(&H1[(size_t)((b0 + row) * 250 + t + 1) * 256 + col], 0, 0);
            }
        }

        // ---- layer 1: weighted skip-sum from bitmask history -> astage ----
        if (tid < 256) {
            const int c = tid;
            float av[16];
#pragma unroll
            for (int i = 0; i < 16; ++i) av[i] = 0.0f;
            for (int k = 0; k < 21; ++k) {
                unsigned w = hist1[((rhead + k) % 21) * 256 + c];
                float wk = w1s[k];
#pragma unroll
                for (int i = 0; i < 16; ++i) av[i] += wk * (float)((w >> i) & 1u);
            }
            const int m = c >> 4, n0 = (c & 15) * 16;
#pragma unroll
            for (int i = 0; i < 16; ++i) astage[m * 256 + n0 + i] = (__bf16)av[i];
        }
        __syncthreads();

        // ---- GEMM1: s_sum @ M1^T (M1 from LDS), LIF layer 1 ----
        v8f acc = {};
#pragma unroll
        for (int k0 = 0; k0 < 256; k0 += 32) {
            v16bf a = load_a_lds(astage, 256, k0, lane);
            v16bf b = load_b_bf16(m1l, 256, wave * 16, k0, lane);
            acc = wmma_bf16(a, b, acc);
        }
        float sp1v[8];
#pragma unroll
        for (int v = 0; v < 8; ++v) {
            int row = v + ((lane >> 4) << 3);
            float h1v = H1[(size_t)((b0 + row) * 250 + t) * 256 + col];
            float vn = 0.5f * v1r[v] + acc[v] + h1v;
            float s  = (vn - 1.0f >= 0.0f) ? 1.0f : 0.0f;
            v1r[v] = vn * (1.0f - s);
            sp1v[v] = s;
        }
        __syncthreads(); // astage reads complete

        if (tid < 16) { rowsum[tid] = 0.0f; rowsumsq[tid] = 0.0f; }
#pragma unroll
        for (int v = 0; v < 8; ++v) {
            int row = v + ((lane >> 4) << 3);
            astage[row * 256 + col] = (__bf16)sp1v[v];
            unsigned mask = __builtin_amdgcn_ballot_w32(sp1v[v] > 0.5f);
            if (lane == 0)  hist1[wslot * 256 + v * 16 + wave]       = (unsigned short)(mask & 0xFFFFu);
            if (lane == 16) hist1[wslot * 256 + (v + 8) * 16 + wave] = (unsigned short)(mask >> 16);
        }
        __syncthreads();

        // ---- GEMM2: spike1 @ W2^T (W2 bf16 from global), layernorm -> h2 ----
        v8f acc2 = {};
#pragma unroll
        for (int k0 = 0; k0 < 256; k0 += 32) {
            v16bf a = load_a_lds(astage, 256, k0, lane);
            v16bf b = load_b_bf16(W2bf, 256, wave * 16, k0, lane);
            acc2 = wmma_bf16(a, b, acc2);
        }
#pragma unroll
        for (int v = 0; v < 8; ++v) {
            float p = acc2[v], q = acc2[v] * acc2[v];
#pragma unroll
            for (int m = 1; m < 16; m <<= 1) {
                p += __shfl_xor(p, m, 32);
                q += __shfl_xor(q, m, 32);
            }
            int row = v + ((lane >> 4) << 3);
            if ((lane & 15) == 0) {
                atomicAdd(&rowsum[row], p);
                atomicAdd(&rowsumsq[row], q);
            }
        }
        __syncthreads();
        float h2[8];
#pragma unroll
        for (int v = 0; v < 8; ++v) {
            int row = v + ((lane >> 4) << 3);
            float mean = rowsum[row] * (1.0f / 256.0f);
            float var  = rowsumsq[row] * (1.0f / 256.0f) - mean * mean;
            h2[v] = (acc2[v] - mean) * rsqrtf(var + 1e-5f) * gg2 + bb2;
        }

        // ---- layer 2: weighted skip-sum -> astage ----
        if (tid < 256) {
            const int c = tid;
            float av[16];
#pragma unroll
            for (int i = 0; i < 16; ++i) av[i] = 0.0f;
            for (int k = 0; k < 21; ++k) {
                unsigned w = hist2[((rhead + k) % 21) * 256 + c];
                float wk = w2s[k];
#pragma unroll
                for (int i = 0; i < 16; ++i) av[i] += wk * (float)((w >> i) & 1u);
            }
            const int m = c >> 4, n0 = (c & 15) * 16;
#pragma unroll
            for (int i = 0; i < 16; ++i) astage[m * 256 + n0 + i] = (__bf16)av[i];
        }
        __syncthreads();

        // ---- GEMM3: s_sum2 @ M2^T (M2 from LDS), LIF layer 2, accumulate ----
        v8f acc3 = {};
#pragma unroll
        for (int k0 = 0; k0 < 256; k0 += 32) {
            v16bf a = load_a_lds(astage, 256, k0, lane);
            v16bf b = load_b_bf16(m2l, 256, wave * 16, k0, lane);
            acc3 = wmma_bf16(a, b, acc3);
        }
        float sp2v[8];
#pragma unroll
        for (int v = 0; v < 8; ++v) {
            float vn = 0.5f * v2r[v] + acc3[v] + h2[v];
            float s  = (vn - 1.0f >= 0.0f) ? 1.0f : 0.0f;
            v2r[v] = vn * (1.0f - s);
            sp2v[v] = s;
            s2r[v] += s;
        }
#pragma unroll
        for (int v = 0; v < 8; ++v) {
            unsigned mask = __builtin_amdgcn_ballot_w32(sp2v[v] > 0.5f);
            if (lane == 0)  hist2[wslot * 256 + v * 16 + wave]       = (unsigned short)(mask & 0xFFFFu);
            if (lane == 16) hist2[wslot * 256 + (v + 8) * 16 + wave] = (unsigned short)(mask >> 16);
        }
        __syncthreads();
    }

    // ---- epilogue: out = (sum_t spike2) @ Wout^T ----
#pragma unroll
    for (int v = 0; v < 8; ++v) {
        int row = v + ((lane >> 4) << 3);
        s2dump[row * 256 + col] = s2r[v];
    }
    __syncthreads();
    if (tid < 320) {
        int m = tid / 20, o = tid % 20;
        float s = 0.0f;
        for (int n = 0; n < 256; ++n) s += s2dump[m * 256 + n] * Wout[o * 256 + n];
        out[(b0 + m) * 20 + o] = s;
    }
}

extern "C" void kernel_launch(void* const* d_in, const int* in_sizes, int n_in,
                              void* d_out, int out_size, void* d_ws, size_t ws_size,
                              hipStream_t stream) {
    const float* x    = (const float*)d_in[0];
    const float* W1   = (const float*)d_in[1];
    const float* g1   = (const float*)d_in[2];
    const float* b1   = (const float*)d_in[3];
    const float* M1   = (const float*)d_in[4];
    const float* sp1  = (const float*)d_in[5];
    const float* W2   = (const float*)d_in[6];
    const float* g2   = (const float*)d_in[7];
    const float* b2   = (const float*)d_in[8];
    const float* M2   = (const float*)d_in[9];
    const float* sp2  = (const float*)d_in[10];
    const float* Wout = (const float*)d_in[11];
    float* out = (float*)d_out;

    // workspace layout
    float*  H1   = (float*)d_ws;                                   // 64000*256 f32 = 65.536 MB
    __bf16* W1bf = (__bf16*)((char*)d_ws + (size_t)64000 * 256 * 4); // 256*704 bf16
    __bf16* W2bf = W1bf + 256 * 704;                               // 256*256 bf16

    prep_weights_kernel<<<352, 512, 0, stream>>>(W1, W2, W1bf, W2bf);
    ff_gemm_ln_kernel<<<4000, 512, 0, stream>>>(x, W1bf, g1, b1, H1);
    scan_kernel<<<16, 512, 2 * 256 * 256 * sizeof(__bf16), stream>>>(
        H1, M1, sp1, W2bf, g2, b2, M2, sp2, Wout, out);
}
